// BeitSelfAttention_55130200212005
// MI455X (gfx1250) — matrix operand order repaired and checked by reading, not compiled, not run on hardware
//
#include <hip/hip_runtime.h>
#include <hip/hip_bf16.h>

#define B_  64
#define S_  197
#define D_  768
#define H_  12
#define HD_ 64
#define SP_ 224      // keys padded to 7*32
#define NT_ 14       // SP_/16 score tiles per q-block
#define KC_ 7        // SP_/32 K-chunks for P*V
#define KSTR 72      // LDS row stride (bf16 elems) for K tile (16B aligned rows)
#define SPV  232     // LDS row stride for transposed V (dims x keys)
#define PSTR 232     // LDS row stride for P staging

typedef __attribute__((ext_vector_type(16))) __bf16 v16bf;
typedef __attribute__((ext_vector_type(8)))  float  v8f;
typedef __attribute__((ext_vector_type(8)))  unsigned v8u;
typedef unsigned short bf16_t;   // storage/interchange type (host-safe)

static __device__ __forceinline__ bf16_t f2bf(float f) {
  unsigned u = __builtin_bit_cast(unsigned, f);
  u += 0x7FFFu + ((u >> 16) & 1u);          // round-to-nearest-even
  return (bf16_t)(u >> 16);
}
static __device__ __forceinline__ unsigned pk2(float a, float b) {
  return (unsigned)f2bf(a) | ((unsigned)f2bf(b) << 16);
}
// combine two 16B chunks into a 16x32-bf16 WMMA operand register set
static __device__ __forceinline__ v16bf mk16(uint4 lo, uint4 hi) {
  v8u t = {lo.x, lo.y, lo.z, lo.w, hi.x, hi.y, hi.z, hi.w};
  return __builtin_bit_cast(v16bf, t);
}

// ---------------------------------------------------------------------------
// Kernel A: f32 -> bf16 bulk convert (8 elements / thread, b128 in, b128 out)
// ---------------------------------------------------------------------------
__global__ void cvt_kernel(const float* __restrict__ src,
                           bf16_t* __restrict__ dst, int n8) {
  int i = blockIdx.x * 256 + threadIdx.x;
  if (i >= n8) return;
  const float4* s = (const float4*)src + (size_t)i * 2;
  float4 a = s[0], b = s[1];
  uint4 o = {pk2(a.x, a.y), pk2(a.z, a.w), pk2(b.x, b.y), pk2(b.z, b.w)};
  ((uint4*)dst)[i] = o;
}

// ---------------------------------------------------------------------------
// Kernel B: materialize relative-position bias  relbias[h][q][k]  (f32)
// ---------------------------------------------------------------------------
__global__ void bias_kernel(const float* __restrict__ bias_table,
                            const int* __restrict__ rel_index,
                            float* __restrict__ relbias) {
  int idx = blockIdx.x * 256 + threadIdx.x;
  if (idx >= S_ * S_) return;
  int q = idx / S_, k = idx % S_;
  int rel = rel_index[idx];
#pragma unroll
  for (int h = 0; h < H_; ++h)
    relbias[((size_t)h * S_ + q) * S_ + k] = bias_table[(size_t)rel * H_ + h];
}

// ---------------------------------------------------------------------------
// Kernel C: QKV projection, LDS-free. grid = (197, 12, 3); block = 64 (2 waves).
// Each wave owns 32 rows x 64 cols (8 accum tiles); B fragments reused by both
// row groups -> 8 WMMA per 12 b128 loads. Stores bf16 head-major qkv[z][b][h][s][d],
// Q pre-scaled by 0.125.
// ---------------------------------------------------------------------------
__global__ void __launch_bounds__(64)
qkv_proj_kernel(const bf16_t* __restrict__ Xb, const bf16_t* __restrict__ Wb,
                const float* __restrict__ bq, const float* __restrict__ bv,
                bf16_t* __restrict__ qkv) {
  const int z = blockIdx.z;
  const bf16_t* W = Wb + (size_t)z * D_ * D_;
  const int tid  = threadIdx.x;
  const int lane = tid & 31, wid = tid >> 5;
  const int lh   = (lane >> 4) & 1, l15 = lane & 15;

  const int m0 = blockIdx.x * 64 + wid * 32;   // this wave's 32 rows
  const int n0 = blockIdx.y * 64;

  const bf16_t* arow0 = Xb + (size_t)(m0 + l15) * D_ + (lh << 3);
  const bf16_t* arow1 = arow0 + (size_t)16 * D_;

  v8f acc[8] = {};
  for (int k0 = 0; k0 < D_; k0 += 32) {
    const uint4* ap0 = (const uint4*)(arow0 + k0);
    const uint4* ap1 = (const uint4*)(arow1 + k0);
    v16bf a0 = mk16(ap0[0], ap0[2]);           // rows m0+l15,    cols +0..7,+16..23
    v16bf a1 = mk16(ap1[0], ap1[2]);           // rows m0+16+l15
#pragma unroll
    for (int c = 0; c < 4; ++c) {
      const uint4* bp =
          (const uint4*)(W + (size_t)(n0 + c * 16 + l15) * D_ + k0 + (lh << 3));
      v16bf bfr = mk16(bp[0], bp[2]);          // K = (lh<<3) + 0..7, +16..23
      acc[c] = __builtin_amdgcn_wmma_f32_16x16x32_bf16(
          false, a0, false, bfr, (short)0, acc[c], false, false);
      acc[4 + c] = __builtin_amdgcn_wmma_f32_16x16x32_bf16(
          false, a1, false, bfr, (short)0, acc[4 + c], false, false);
    }
  }

  const float scale = (z == 0) ? 0.125f : 1.0f;   // 1/sqrt(64) folded into Q
#pragma unroll
  for (int c = 0; c < 4; ++c) {
    int n = n0 + c * 16 + l15;
    float bias = (z == 0) ? bq[n] : (z == 2) ? bv[n] : 0.0f;
    int h = n >> 6, d = n & 63;
#pragma unroll
    for (int g = 0; g < 2; ++g) {               // two 16-row groups
#pragma unroll
      for (int r = 0; r < 8; ++r) {
        int m = m0 + g * 16 + r + (lh << 3);
        int b = m / S_, s = m % S_;
        size_t dst = ((((size_t)z * B_ + b) * H_ + h) * S_ + s) * HD_ + d;
        qkv[dst] = f2bf((acc[g * 4 + c][r] + bias) * scale);
      }
    }
  }
}

// ---------------------------------------------------------------------------
// Kernel D: fused attention per (b,h).  grid = B*H, block = 128 (4 waves).
// K staged via global_load_async_to_lds_b128 (ASYNCcnt); V staged transposed.
// ---------------------------------------------------------------------------
__global__ void __launch_bounds__(128)
attn_kernel(const bf16_t* __restrict__ qkv, const float* __restrict__ relbias,
            float* __restrict__ out) {
  const int bh = blockIdx.x;
  const int b = bh / H_, h = bh % H_;
  const int tid  = threadIdx.x;
  const int lane = tid & 31, wid = tid >> 5;
  const int lh   = (lane >> 4) & 1, l15 = lane & 15;

  __shared__ bf16_t Ks[SP_ * KSTR];      // [key][dim]   (pad rows uninit: masked)
  __shared__ bf16_t Vt[HD_ * SPV];       // [dim][key]   (pad keys zeroed)
  __shared__ bf16_t Ps[4][16 * PSTR];    // per-wave P staging

  const size_t qbase = (((size_t)0 * B_ + b) * H_ + h) * S_ * HD_;
  const size_t kbase = (((size_t)1 * B_ + b) * H_ + h) * S_ * HD_;
  const size_t vbase = (((size_t)2 * B_ + b) * H_ + h) * S_ * HD_;

  // ---- K tile: async DMA global -> LDS, 16B per lane-op (ASYNCcnt) -------
  for (int i = tid; i < S_ * 8; i += 128) {          // 197 rows x 8 chunks
    int row = i >> 3, c8 = (i & 7) << 3;
    unsigned lds_k = (unsigned)(unsigned long long)&Ks[row * KSTR + c8];
    unsigned long long gk =
        (unsigned long long)(qkv + kbase + (size_t)row * HD_ + c8);
    asm volatile("global_load_async_to_lds_b128 %0, %1, off"
                 :: "v"(lds_k), "v"(gk) : "memory");
  }

  // ---- V tile: load rows coalesced, scatter transposed into Vt -----------
  for (int i = tid; i < SP_ * 8; i += 128) {         // 224 keys x 8 chunks
    int key = i >> 3, d8 = (i & 7) << 3;
    uint4 val = {0u, 0u, 0u, 0u};
    if (key < S_) val = *(const uint4*)(qkv + vbase + (size_t)key * HD_ + d8);
    unsigned u[4] = {val.x, val.y, val.z, val.w};
#pragma unroll
    for (int j = 0; j < 4; ++j) {
      Vt[(d8 + 2 * j + 0) * SPV + key] = (bf16_t)(u[j] & 0xffffu);
      Vt[(d8 + 2 * j + 1) * SPV + key] = (bf16_t)(u[j] >> 16);
    }
  }

  asm volatile("s_wait_asynccnt 0" ::: "memory");
  __syncthreads();

  for (int qb = wid; qb < 13; qb += 4) {             // wave-uniform loop
    // ---- Q fragments: 16x64, two K-chunks, b128 loads --------------------
    int qrow = qb * 16 + l15; if (qrow > S_ - 1) qrow = S_ - 1;
    const uint4* qp = (const uint4*)(qkv + qbase + (size_t)qrow * HD_ + (lh << 4));
    v16bf a0 = mk16(qp[0], qp[1]);
    v16bf a1 = mk16(qp[4], qp[5]);

    // ---- scores = Q K^T : 14 tiles x 2 wmma ------------------------------
    v8f sc[NT_];
#pragma unroll
    for (int t = 0; t < NT_; ++t) {
      int key = t * 16 + l15;
      const uint4* kp0 = (const uint4*)(Ks + key * KSTR + (lh << 4));
      const uint4* kp1 = (const uint4*)(Ks + key * KSTR + 32 + (lh << 4));
      v16bf b0 = mk16(kp0[0], kp0[1]);
      v16bf b1 = mk16(kp1[0], kp1[1]);
      v8f acc = {};
      acc = __builtin_amdgcn_wmma_f32_16x16x32_bf16(
          false, a0, false, b0, (short)0, acc, false, false);
      acc = __builtin_amdgcn_wmma_f32_16x16x32_bf16(
          false, a1, false, b1, (short)0, acc, false, false);
      sc[t] = acc;
    }

    // ---- add relative-position bias, mask padded keys --------------------
#pragma unroll
    for (int t = 0; t < NT_; ++t) {
#pragma unroll
      for (int r = 0; r < 8; ++r) {
        int q = qb * 16 + r + (lh << 3);
        int k = t * 16 + l15;
        float v = sc[t][r];
        if (k < S_) {
          int qc = (q < S_) ? q : S_ - 1;
          v += relbias[((size_t)h * S_ + qc) * S_ + k];
        } else {
          v = -1e30f;                     // overwrites any garbage from pad rows
        }
        sc[t][r] = v;
      }
    }

    // ---- softmax: each row lives in one 16-lane half ---------------------
    float inv[8];
#pragma unroll
    for (int r = 0; r < 8; ++r) {
      float mx = -1e30f;
#pragma unroll
      for (int t = 0; t < NT_; ++t) mx = fmaxf(mx, sc[t][r]);
      mx = fmaxf(mx, __shfl_xor(mx, 1));
      mx = fmaxf(mx, __shfl_xor(mx, 2));
      mx = fmaxf(mx, __shfl_xor(mx, 4));
      mx = fmaxf(mx, __shfl_xor(mx, 8));
      float sum = 0.0f;
#pragma unroll
      for (int t = 0; t < NT_; ++t) {
        float e = __expf(sc[t][r] - mx);
        sc[t][r] = e;
        sum += e;
      }
      sum += __shfl_xor(sum, 1);
      sum += __shfl_xor(sum, 2);
      sum += __shfl_xor(sum, 4);
      sum += __shfl_xor(sum, 8);
      inv[r] = 1.0f / sum;
    }

    // ---- stage P (bf16) through LDS: C/D-layout -> row-major -------------
#pragma unroll
    for (int t = 0; t < NT_; ++t)
#pragma unroll
      for (int r = 0; r < 8; ++r)
        Ps[wid][(r + (lh << 3)) * PSTR + t * 16 + l15] = f2bf(sc[t][r] * inv[r]);
    // same-wave LDS ops are in-order; no barrier needed

    // ---- ctx = P V : 7 K-chunks x 4 tiles, all-b128 LDS reads ------------
    v8f ctx[4] = {};
#pragma unroll
    for (int kt = 0; kt < KC_; ++kt) {
      const uint4* pp = (const uint4*)(Ps[wid] + l15 * PSTR + kt * 32 + (lh << 4));
      v16bf pa = mk16(pp[0], pp[1]);
#pragma unroll
      for (int c = 0; c < 4; ++c) {
        const uint4* vp =
            (const uint4*)(Vt + (c * 16 + l15) * SPV + kt * 32 + (lh << 4));
        v16bf vb = mk16(vp[0], vp[1]);
        ctx[c] = __builtin_amdgcn_wmma_f32_16x16x32_bf16(
            false, pa, false, vb, (short)0, ctx[c], false, false);
      }
    }

    // ---- store ctx -> out[b][q][h*64 + d] (f32) --------------------------
#pragma unroll
    for (int c = 0; c < 4; ++c)
#pragma unroll
      for (int r = 0; r < 8; ++r) {
        int q = qb * 16 + r + (lh << 3);
        if (q < S_)
          out[((size_t)b * S_ + q) * D_ + h * HD_ + c * 16 + l15] = ctx[c][r];
      }
  }
}

// ---------------------------------------------------------------------------
extern "C" void kernel_launch(void* const* d_in, const int* in_sizes, int n_in,
                              void* d_out, int out_size, void* d_ws, size_t ws_size,
                              hipStream_t stream) {
  const float* hs         = (const float*)d_in[0];
  const float* wq         = (const float*)d_in[1];
  const float* bq         = (const float*)d_in[2];
  const float* wk         = (const float*)d_in[3];
  const float* wv         = (const float*)d_in[4];
  const float* bv         = (const float*)d_in[5];
  const float* bias_table = (const float*)d_in[6];
  const int*   rel_index  = (const int*)d_in[7];
  float* out = (float*)d_out;

  // workspace: [Xbf 19.4MB][Wbf 3.5MB][qkv 58.1MB][relbias 1.9MB]  (~83MB)
  char* ws = (char*)d_ws;
  bf16_t* Xb  = (bf16_t*)ws;                 ws += (size_t)B_ * S_ * D_ * 2;
  bf16_t* Wb  = (bf16_t*)ws;                 ws += (size_t)3 * D_ * D_ * 2;
  bf16_t* qkv = (bf16_t*)ws;                 ws += (size_t)3 * B_ * H_ * S_ * HD_ * 2;
  float* relbias = (float*)ws;

  const int XN8 = B_ * S_ * D_ / 8;          // 1,210,368
  const int WN8 = D_ * D_ / 8;               // 73,728

  cvt_kernel<<<(XN8 + 255) / 256, 256, 0, stream>>>(hs, Xb, XN8);
  cvt_kernel<<<(WN8 + 255) / 256, 256, 0, stream>>>(wq, Wb + (size_t)0 * D_ * D_, WN8);
  cvt_kernel<<<(WN8 + 255) / 256, 256, 0, stream>>>(wk, Wb + (size_t)1 * D_ * D_, WN8);
  cvt_kernel<<<(WN8 + 255) / 256, 256, 0, stream>>>(wv, Wb + (size_t)2 * D_ * D_, WN8);
  bias_kernel<<<(S_ * S_ + 255) / 256, 256, 0, stream>>>(bias_table, rel_index, relbias);
  qkv_proj_kernel<<<dim3(197, 12, 3), 64, 0, stream>>>(Xb, Wb, bq, bv, qkv);
  attn_kernel<<<dim3(B_ * H_), 128, 0, stream>>>(qkv, relbias, out);
}
